// DGCNNCaps_part_86320252715538
// MI455X (gfx1250) — compile-verified, matrix operand order checked by reading, not run on hardware
//
#include <hip/hip_runtime.h>
#include <hip/hip_bf16.h>

// ---------------------------------------------------------------------------
// DGCNN-Caps forward for MI455X (gfx1250, wave32).
// All GEMM-shaped einsums (edgeconv, linear encoder, capsule projection,
// latent embedding, decoder convs) run through V_WMMA_F32_16X16X32_F16
// (f16 inputs, f32 accumulate). kNN/top-k, BN/LN stats, routing softmax,
// squash run in fp32 VALU. Activations are stored point-major [L][K] so each
// WMMA fragment is two contiguous 128-bit loads per lane. Per-lane fragment
// base pointers are hoisted out of the k-loops (only a uniform +k0 offset
// remains -> immediate offsets after unroll, no spills), ragged-M weights are
// row-padded so loads are unguarded, and GEMM waves compute 16x64 tiles
// (A reuse, 5 loads / 4 wmma) with software pipelining + global_prefetch.
// ---------------------------------------------------------------------------

typedef __attribute__((ext_vector_type(16))) _Float16 v16h;
typedef __attribute__((ext_vector_type(8)))  _Float16 v8h;
typedef __attribute__((ext_vector_type(8)))  float    v8f;

#define DEV_INLINE __device__ __forceinline__

static constexpr int B_   = 32;
static constexpr int NPTS = 1024;
static constexpr int KNNK = 20;

// --------------------------- WMMA helpers ----------------------------------
// 16-bit A (16x32) / B (32x16) fragment layout (ISA 7.12.2):
//   lane = {r,h}: r = lane&15 (row for A / col for B), h = lane>>4
//   half j in [0,16):  k = 16*(j>>3) + 8*h + (j&7)
// => two contiguous 8-half (128-bit) loads per fragment, unguarded.
DEV_INLINE const _Float16* frag_ptr(const _Float16* base, int r0, int ld) {
  int lane = (int)(threadIdx.x & 31u);
  return base + (size_t)(r0 + (lane & 15)) * (size_t)ld + 8 * (lane >> 4);
}

DEV_INLINE v16h load_frag_at(const _Float16* p) {
  v8h lo = *(const v8h*)(p);
  v8h hi = *(const v8h*)(p + 16);
  v16h f;
#pragma unroll
  for (int j = 0; j < 8; ++j) { f[j] = lo[j]; f[j + 8] = hi[j]; }
  return f;
}

DEV_INLINE v8f wmma_f16(v16h a, v16h b, v8f c) {
  return __builtin_amdgcn_wmma_f32_16x16x32_f16(false, a, false, b, (short)0, c,
                                                false, false);
}

// order-preserving encoding of float into uint (for atomic max/min)
DEV_INLINE unsigned encf(float f) {
  unsigned u = __float_as_uint(f);
  return (u & 0x80000000u) ? ~u : (u | 0x80000000u);
}
DEV_INLINE float decf(unsigned u) {
  return __uint_as_float((u & 0x80000000u) ? (u & 0x7fffffffu) : ~u);
}

// --------------------------- generic GEMM ----------------------------------
// out[b][l][m] = sum_k A[b][m][k] * Bt[b][l][k]   (A row-major MxK, Bt LxK)
// block = 128 threads (4 waves); each wave owns a 16M x 64L tile:
// one A fragment is reused across 4 B fragments (5 loads per 4 wmma),
// software-pipelined one k-step ahead + global prefetch.
// A must have rows padded to a multiple of 16 (store guard uses M).
__global__ void __launch_bounds__(128, 1)
gemm_f16_kernel(const _Float16* __restrict__ A, size_t a_bs,
                const _Float16* __restrict__ Bt, size_t b_bs,
                float* __restrict__ out, size_t o_bs,
                int M, int K, int L, int out_ld) {
  int b = blockIdx.z;
  int wave = (int)(threadIdx.x >> 5u);
  int lane = (int)(threadIdx.x & 31u);
  int col0 = (blockIdx.x * 4 + wave) * 64;
  int m0 = blockIdx.y * 16;
  const _Float16* pa  = frag_ptr(A + a_bs * b, m0, K);
  const _Float16* pb0 = frag_ptr(Bt + b_bs * b, col0, K);
  const _Float16* pb1 = pb0 + (size_t)16 * K;
  const _Float16* pb2 = pb0 + (size_t)32 * K;
  const _Float16* pb3 = pb0 + (size_t)48 * K;
  v8f acc0 = {}, acc1 = {}, acc2 = {}, acc3 = {};
  v16h a0 = load_frag_at(pa);
  v16h b0 = load_frag_at(pb0);
  v16h b1 = load_frag_at(pb1);
  v16h b2 = load_frag_at(pb2);
  v16h b3 = load_frag_at(pb3);
  for (int k0 = 32; k0 < K; k0 += 32) {
    if (k0 + 32 < K) {
      __builtin_prefetch(pa + k0 + 32, 0, 1);
      __builtin_prefetch(pb0 + k0 + 32, 0, 1);
    }
    v16h an  = load_frag_at(pa + k0);
    v16h bn0 = load_frag_at(pb0 + k0);
    v16h bn1 = load_frag_at(pb1 + k0);
    v16h bn2 = load_frag_at(pb2 + k0);
    v16h bn3 = load_frag_at(pb3 + k0);
    acc0 = wmma_f16(a0, b0, acc0);
    acc1 = wmma_f16(a0, b1, acc1);
    acc2 = wmma_f16(a0, b2, acc2);
    acc3 = wmma_f16(a0, b3, acc3);
    a0 = an; b0 = bn0; b1 = bn1; b2 = bn2; b3 = bn3;
  }
  acc0 = wmma_f16(a0, b0, acc0);
  acc1 = wmma_f16(a0, b1, acc1);
  acc2 = wmma_f16(a0, b2, acc2);
  acc3 = wmma_f16(a0, b3, acc3);

  int n = lane & 15, h = lane >> 4;
  v8f accs[4] = {acc0, acc1, acc2, acc3};
#pragma unroll
  for (int t = 0; t < 4; ++t) {
    float* orow = out + o_bs * b + (size_t)(col0 + 16 * t + n) * out_ld;
#pragma unroll
    for (int i = 0; i < 8; ++i) {
      int m = m0 + 8 * h + i;
      if (m < M) orow[m] = accs[t][i];  // 8 consecutive floats per lane
    }
  }
}

// --------------------------- kNN (top-20) ----------------------------------
// block = 256 threads = 8 waves; wave owns one query row; 64-col LDS tiles.
template <int C>
__global__ void __launch_bounds__(256, 1)
knn_kernel(const float* __restrict__ x, int* __restrict__ idxout) {
  __shared__ float Xt[C * 64];
  __shared__ float cn[64];
  __shared__ float xi[8 * C];
  __shared__ float md[8 * 640];
  __shared__ int   mi[8 * 640];
  const int tid = (int)threadIdx.x;
  const int wave = tid >> 5, lane = tid & 31;
  const int b = blockIdx.y;
  const int n0 = blockIdx.x * 8;
  const int n = n0 + wave;

  for (int t = tid; t < 8 * C; t += 256) {
    int row = t / C, c = t % C;
    xi[row * C + c] = x[((size_t)b * C + c) * NPTS + n0 + row];
  }
  __syncthreads();
  float xi2 = 0.f;
  for (int c = lane; c < C; c += 32) { float v = xi[wave * C + c]; xi2 += v * v; }
#pragma unroll
  for (int off = 16; off; off >>= 1) xi2 += __shfl_xor(xi2, off);

  float bd[20]; int bi[20];
#pragma unroll
  for (int s = 0; s < 20; ++s) { bd[s] = -3.4e38f; bi[s] = 0; }
  float minv = -3.4e38f; int minslot = 0;

  for (int ct = 0; ct < NPTS / 64; ++ct) {
    __syncthreads();
    for (int t = tid; t < C * 64; t += 256) {
      int c = t >> 6, j = t & 63;
      Xt[c * 64 + j] = x[((size_t)b * C + c) * NPTS + ct * 64 + j];
    }
    __syncthreads();
    if (tid < 64) {
      float s = 0.f;
      for (int c = 0; c < C; ++c) { float v = Xt[c * 64 + tid]; s += v * v; }
      cn[tid] = s;
    }
    __syncthreads();
    for (int jj = lane; jj < 64; jj += 32) {
      float dot = 0.f;
      for (int c = 0; c < C; ++c) dot += xi[wave * C + c] * Xt[c * 64 + jj];
      float pd = 2.f * dot - xi2 - cn[jj];
      int col = ct * 64 + jj;
      if (pd > minv) {
#pragma unroll
        for (int s = 0; s < 20; ++s) if (s == minslot) { bd[s] = pd; bi[s] = col; }
        minv = bd[0]; minslot = 0;
#pragma unroll
        for (int s = 1; s < 20; ++s) if (bd[s] < minv) { minv = bd[s]; minslot = s; }
      }
    }
  }
  __syncthreads();
#pragma unroll
  for (int s = 0; s < 20; ++s) {
    md[wave * 640 + lane * 20 + s] = bd[s];
    mi[wave * 640 + lane * 20 + s] = bi[s];
  }
  __syncthreads();
  // cooperative top-20 selection over the 32x20 merged candidates
  for (int t = 0; t < 20; ++t) {
    float lv = -3.4e38f; int ls = 0;
#pragma unroll
    for (int s = 0; s < 20; ++s) {
      float v = md[wave * 640 + lane * 20 + s];
      if (v > lv) { lv = v; ls = s; }
    }
    int lcol = mi[wave * 640 + lane * 20 + ls];
    int lln = lane;
#pragma unroll
    for (int off = 16; off; off >>= 1) {
      float ov = __shfl_xor(lv, off);
      int oc = __shfl_xor(lcol, off);
      int ol = __shfl_xor(lln, off);
      if (ov > lv || (ov == lv && ol < lln)) { lv = ov; lcol = oc; lln = ol; }
    }
    if (lane == lln) md[wave * 640 + lane * 20 + ls] = -3.4e38f;
    if (lane == t) idxout[((size_t)b * NPTS + n) * KNNK + t] = lcol;
    __syncthreads();
  }
}

// --------------------------- EdgeConv --------------------------------------
// Gather edge features into LDS (point-major, f16); per wave: one B fragment
// (edge tile) reused across all COUT/16 accumulators; per-(b,o,n) max/min
// over k via encoded atomics; per-channel sums via half-wave shfl reduce.
template <int CINR, int KP, int COUT>
__global__ void __launch_bounds__(128, 1)
edgeconv_kernel(const float* __restrict__ x, const int* __restrict__ idx,
                const _Float16* __restrict__ W,
                unsigned* __restrict__ ymax, unsigned* __restrict__ ymin,
                float* __restrict__ stats) {
  __shared__ _Float16 Et[64 * KP];
  __shared__ int jj[64];
  __shared__ float lsum[COUT];
  __shared__ float lssq[COUT];
  const int tid = (int)threadIdx.x;
  const int b = blockIdx.y;
  const int c0 = blockIdx.x * 64;
  if (tid < COUT) { lsum[tid] = 0.f; lssq[tid] = 0.f; }
  if (tid < 64) {
    int flat = c0 + tid;
    jj[tid] = idx[((size_t)b * NPTS + flat / KNNK) * KNNK + flat % KNNK];
  }
  __syncthreads();
  for (int t = tid; t < 64 * KP; t += 128) {
    int col = t / KP, kk = t % KP;
    int flat = c0 + col;
    int n = flat / KNNK;
    float val = 0.f;
    if (kk < CINR) {
      int j = jj[col];
      val = x[((size_t)b * CINR + kk) * NPTS + j] -
            x[((size_t)b * CINR + kk) * NPTS + n];
    } else if (kk < 2 * CINR) {
      val = x[((size_t)b * CINR + (kk - CINR)) * NPTS + n];
    }
    Et[col * KP + kk] = (_Float16)val;
  }
  __syncthreads();
  const int wave = tid >> 5, lane = tid & 31;
  const int wcol = wave * 16;
  const int nlo = lane & 15, h = lane >> 4;
  const int flat = c0 + wcol + nlo;
  const int n = flat / KNNK;
  constexpr int MT = COUT / 16;
  const _Float16* pw = frag_ptr(W, 0, KP);
  const _Float16* pe = frag_ptr(Et, wcol, KP);
  v8f acc[MT];
  v8f zero = {};
#pragma unroll
  for (int mt = 0; mt < MT; ++mt) acc[mt] = zero;
#pragma unroll
  for (int k0 = 0; k0 < KP; k0 += 32) {
    v16h bf = load_frag_at(pe + k0);
#pragma unroll
    for (int mt = 0; mt < MT; ++mt) {
      v16h af = load_frag_at(pw + mt * 16 * KP + k0);
      acc[mt] = wmma_f16(af, bf, acc[mt]);
    }
  }
#pragma unroll
  for (int mt = 0; mt < MT; ++mt) {
#pragma unroll
    for (int i = 0; i < 8; ++i) {
      int o = mt * 16 + 8 * h + i;
      float y = acc[mt][i];
      unsigned e = encf(y);
      atomicMax(&ymax[((size_t)b * COUT + o) * NPTS + n], e);
      atomicMin(&ymin[((size_t)b * COUT + o) * NPTS + n], e);
      float r1 = y, r2 = y * y;
#pragma unroll
      for (int off = 8; off; off >>= 1) { r1 += __shfl_xor(r1, off); r2 += __shfl_xor(r2, off); }
      if (nlo == 0) { atomicAdd(&lsum[o], r1); atomicAdd(&lssq[o], r2); }
    }
  }
  __syncthreads();
  if (tid < COUT) {
    atomicAdd(&stats[tid], lsum[tid]);
    atomicAdd(&stats[COUT + tid], lssq[tid]);
  }
}

// BN (batch stats) + LeakyReLU + max-over-k (monotone => use max or min per sign(g))
__global__ void bn_max_kernel(const unsigned* __restrict__ ymax, const unsigned* __restrict__ ymin,
                              const float* __restrict__ stats, const float* __restrict__ g,
                              const float* __restrict__ bb, float* __restrict__ xout,
                              int COUT, float invcnt) {
  size_t i = (size_t)blockIdx.x * blockDim.x + threadIdx.x;
  size_t tot = (size_t)B_ * COUT * NPTS;
  if (i >= tot) return;
  int o = (int)((i / NPTS) % COUT);
  float mean = stats[o] * invcnt;
  float var = stats[COUT + o] * invcnt - mean * mean;
  float gv = g[o];
  float val = decf(gv >= 0.f ? ymax[i] : ymin[i]);
  float t = gv * (val - mean) * rsqrtf(var + 1e-5f) + bb[o];
  xout[i] = t >= 0.f ? t : 0.2f * t;
}

// --------------------------- capsule projection ----------------------------
// u_hat[b,l,p,v] = sum_d enc[b,p,d] * caps_W[l,p,v,d]
// per (l,p): M=32 (batch), K=48 (d padded), Ncols=64 (v). One block (4 waves),
// B fragment reused across both batch m-tiles.
__global__ void __launch_bounds__(128, 1)
uhat_kernel(const _Float16* __restrict__ enc16,
            const _Float16* __restrict__ cw16,
            float* __restrict__ uhat) {
  const int p = blockIdx.x;
  const int l = blockIdx.y;
  const int wave = (int)(threadIdx.x >> 5u), lane = (int)(threadIdx.x & 31u);
  const int v0 = wave * 16;
  const _Float16* pa = frag_ptr(enc16 + (size_t)p * 64, 0, NPTS * 64);  // rows = batch
  const _Float16* pb = frag_ptr(cw16 + ((size_t)l * NPTS + p) * (64 * 48), v0, 48);
  const int nlo = lane & 15, h = lane >> 4;
  v8f acc0 = {}, acc1 = {};
#pragma unroll
  for (int k0 = 0; k0 < 64; k0 += 32) {
    // ld=48: reads past row tail hit the next (finite) f16 row; the matching
    // A halves (k>=40) are zero-padded so those products contribute 0.
    v16h bf = load_frag_at(pb + k0);
    v16h a0 = load_frag_at(pa + k0);
    v16h a1 = load_frag_at(pa + (size_t)16 * NPTS * 64 + k0);
    acc0 = wmma_f16(a0, bf, acc0);
    acc1 = wmma_f16(a1, bf, acc1);
  }
#pragma unroll
  for (int i = 0; i < 8; ++i) {
    int bidx = 8 * h + i;
    uhat[(((size_t)bidx * 40 + l) * NPTS + p) * 64 + v0 + nlo] = acc0[i];
    uhat[((((size_t)bidx + 16) * 40 + l) * NPTS + p) * 64 + v0 + nlo] = acc1[i];
  }
}

// --------------------------- small VALU kernels ----------------------------
__global__ void fill_u32_kernel(unsigned* p, unsigned v, size_t n) {
  size_t i = (size_t)blockIdx.x * blockDim.x + threadIdx.x;
  if (i < n) p[i] = v;
}
__global__ void fill_f32_kernel(float* p, float v, size_t n) {
  size_t i = (size_t)blockIdx.x * blockDim.x + threadIdx.x;
  if (i < n) p[i] = v;
}
// f32 -> f16 with row and column zero-padding (rows R->Rp, cols Kin->Kp)
__global__ void cvt_pad_kernel(const float* __restrict__ src, _Float16* __restrict__ dst,
                               int R, int Rp, int Kin, int Kp) {
  size_t i = (size_t)blockIdx.x * blockDim.x + threadIdx.x;
  size_t tot = (size_t)Rp * Kp;
  if (i >= tot) return;
  int k = (int)(i % Kp);
  size_t r = i / Kp;
  dst[i] = (_Float16)((r < (size_t)R && k < Kin) ? src[r * Kin + k] : 0.f);
}
__global__ void feat_kernel(const float* __restrict__ x1, const float* __restrict__ x2,
                            const float* __restrict__ x3, _Float16* __restrict__ feat16) {
  size_t i = (size_t)blockIdx.x * blockDim.x + threadIdx.x;
  size_t tot = (size_t)B_ * NPTS * 256;
  if (i >= tot) return;
  int c = (int)(i % 256);
  size_t bn = i / 256;
  int n = (int)(bn % NPTS);
  int b = (int)(bn / NPTS);
  float v;
  if (c < 64)       v = x1[((size_t)b * 64 + c) * NPTS + n];
  else if (c < 128) v = x2[((size_t)b * 64 + (c - 64)) * NPTS + n];
  else              v = x3[((size_t)b * 128 + (c - 128)) * NPTS + n];
  feat16[i] = (_Float16)v;
}
__global__ void ln_stats_kernel(const float* __restrict__ h, float* __restrict__ lnst) {
  int b = blockIdx.x;
  float s = 0.f, s2 = 0.f;
  for (int i = (int)threadIdx.x; i < NPTS * 40; i += 256) {
    float v = h[(size_t)b * NPTS * 40 + i];
    s += v; s2 += v * v;
  }
  __shared__ float rs[256], rq[256];
  rs[threadIdx.x] = s; rq[threadIdx.x] = s2; __syncthreads();
  for (int off = 128; off; off >>= 1) {
    if ((int)threadIdx.x < off) {
      rs[threadIdx.x] += rs[threadIdx.x + off];
      rq[threadIdx.x] += rq[threadIdx.x + off];
    }
    __syncthreads();
  }
  if (threadIdx.x == 0) {
    float mean = rs[0] / (NPTS * 40.f);
    lnst[b * 2] = mean;
    lnst[b * 2 + 1] = rq[0] / (NPTS * 40.f) - mean * mean;
  }
}
__global__ void enc_kernel(const float* __restrict__ h, const float* __restrict__ lnw,
                           const float* __restrict__ lnb, const float* __restrict__ lnst,
                           float* __restrict__ enc32, _Float16* __restrict__ enc16) {
  size_t i = (size_t)blockIdx.x * blockDim.x + threadIdx.x;
  size_t tot = (size_t)B_ * NPTS * 64;
  if (i >= tot) return;
  int d = (int)(i % 64);
  size_t pn = i / 64;
  int n = (int)(pn % NPTS);
  int b = (int)(pn / NPTS);
  float e = 0.f;
  if (d < 40) {
    float v = h[((size_t)b * NPTS + n) * 40 + d];
    float mu = lnst[b * 2], var = lnst[b * 2 + 1];
    e = lnw[n * 40 + d] * (v - mu) * rsqrtf(var + 1e-6f) + lnb[n * 40 + d];
    e = e > 0.f ? e : 0.f;
    enc32[((size_t)b * NPTS + n) * 40 + d] = e;
  }
  enc16[i] = (_Float16)e;
}
__global__ void softmax_l_kernel(const float* __restrict__ bij, float* __restrict__ cc) {
  int i = blockIdx.x * blockDim.x + threadIdx.x;
  if (i >= B_ * NPTS) return;
  int b = i / NPTS, p = i % NPTS;
  const float* src = bij + (size_t)b * 40 * NPTS + p;
  float mx = -3.4e38f;
  for (int l = 0; l < 40; ++l) mx = fmaxf(mx, src[(size_t)l * NPTS]);
  float s = 0.f;
  for (int l = 0; l < 40; ++l) s += expf(src[(size_t)l * NPTS] - mx);
  float inv = 1.f / s;
  float* dst = cc + (size_t)b * 40 * NPTS + p;
  for (int l = 0; l < 40; ++l) dst[(size_t)l * NPTS] = expf(src[(size_t)l * NPTS] - mx) * inv;
}
__global__ void sv_squash_kernel(const float* __restrict__ cc, const float* __restrict__ uhat,
                                 float* __restrict__ vout) {
  int l = blockIdx.x, b = blockIdx.y;
  int tid = (int)threadIdx.x;
  int v = tid & 63, pg = tid >> 6;
  float partial = 0.f;
  const float* crow = cc + ((size_t)b * 40 + l) * NPTS;
  const float* urow = uhat + ((size_t)b * 40 + l) * NPTS * 64;
  for (int p = pg; p < NPTS; p += 4) partial += crow[p] * urow[(size_t)p * 64 + v];
  __shared__ float red[256];
  __shared__ float svv[64];
  __shared__ float snv;
  red[tid] = partial; __syncthreads();
  if (tid < 64) svv[tid] = red[tid] + red[tid + 64] + red[tid + 128] + red[tid + 192];
  __syncthreads();
  if (tid == 0) { float sn = 0.f; for (int k = 0; k < 64; ++k) sn += svv[k] * svv[k]; snv = sn; }
  __syncthreads();
  if (tid < 64) {
    float sn = snv;
    vout[((size_t)b * 40 + l) * 64 + tid] = svv[tid] * sn / ((1.f + sn) * sqrtf(sn));
  }
}
__global__ void bij_update_kernel(const float* __restrict__ vout, const float* __restrict__ uhat,
                                  float* __restrict__ bij) {
  int i = blockIdx.x * blockDim.x + threadIdx.x;
  if (i >= B_ * 40 * NPTS) return;
  int l = (i / NPTS) % 40;
  int b = i / (40 * NPTS);
  const float* vv = vout + ((size_t)b * 40 + l) * 64;
  const float* uu = uhat + (size_t)i * 64;
  float dot = 0.f;
  for (int k = 0; k < 64; ++k) dot += vv[k] * uu[k];
  bij[i] += dot;
}
__global__ void latT_kernel(const float* __restrict__ vout, _Float16* __restrict__ latT) {
  int i = blockIdx.x * blockDim.x + threadIdx.x;
  if (i >= B_ * 64 * 64) return;
  int k = i % 64;
  int m = (i / 64) % 64;
  int b = i / 4096;
  latT[i] = (_Float16)(k < 40 ? vout[((size_t)b * 40 + k) * 64 + m] : 0.f);
}
__global__ void embmax_kernel(const float* __restrict__ embf, float* __restrict__ embmax) {
  int i = blockIdx.x * blockDim.x + threadIdx.x;
  if (i >= B_ * NPTS) return;
  const float* row = embf + (size_t)i * 64;
  float m = row[0];
  for (int v = 1; v < 64; ++v) m = fmaxf(m, row[v]);
  embmax[i] = m;
}
__global__ void lab_kernel(const float* __restrict__ lin, const float* __restrict__ cw1,
                           const float* __restrict__ cg1, const float* __restrict__ cb1,
                           float* __restrict__ labact) {
  int o = (int)threadIdx.x;  // 64 threads
  float ys[32];
  for (int b = 0; b < 32; ++b) {
    float s = 0.f;
    for (int c = 0; c < 16; ++c) s += cw1[o * 16 + c] * lin[b * 16 + c];
    ys[b] = s;
  }
  float m = 0.f;
  for (int b = 0; b < 32; ++b) m += ys[b];
  m *= (1.f / 32.f);
  float v = 0.f;
  for (int b = 0; b < 32; ++b) { float d = ys[b] - m; v += d * d; }
  v *= (1.f / 32.f);
  float r = rsqrtf(v + 1e-5f), gg = cg1[o], bb = cb1[o];
  for (int b = 0; b < 32; ++b) {
    float t = gg * (ys[b] - m) * r + bb;
    labact[b * 64 + o] = t >= 0.f ? t : 0.2f * t;
  }
}
__global__ void embfull_kernel(const float* __restrict__ embmax, const float* __restrict__ labact,
                               const float* __restrict__ enc32, _Float16* __restrict__ ef) {
  size_t i = (size_t)blockIdx.x * blockDim.x + threadIdx.x;
  size_t tot = (size_t)B_ * NPTS * 1152;
  if (i >= tot) return;
  int ch = (int)(i % 1152);
  size_t bn = i / 1152;
  int n = (int)(bn % NPTS);
  int b = (int)(bn / NPTS);
  float v = 0.f;
  if (ch < 1024)      v = embmax[(size_t)b * NPTS + ch];
  else if (ch < 1088) v = labact[b * 64 + (ch - 1024)];
  else if (ch < 1128) v = enc32[((size_t)b * NPTS + n) * 40 + (ch - 1088)];
  ef[i] = (_Float16)v;
}
// deterministic per-channel sum / sumsq (one block per channel)
__global__ void chan_stats_kernel(const float* __restrict__ y, int ld, int O, int rows,
                                  float* __restrict__ stats) {
  int o = blockIdx.x;
  float s = 0.f, s2 = 0.f;
  for (int r = (int)threadIdx.x; r < rows; r += 256) {
    float v = y[(size_t)r * ld + o];
    s += v; s2 += v * v;
  }
  __shared__ float rs[256], rq[256];
  rs[threadIdx.x] = s; rq[threadIdx.x] = s2; __syncthreads();
  for (int off = 128; off; off >>= 1) {
    if ((int)threadIdx.x < off) {
      rs[threadIdx.x] += rs[threadIdx.x + off];
      rq[threadIdx.x] += rq[threadIdx.x + off];
    }
    __syncthreads();
  }
  if (threadIdx.x == 0) { stats[o] = rs[0]; stats[O + o] = rq[0]; }
}
__global__ void bn_act_f16_kernel(const float* __restrict__ y, const float* __restrict__ g,
                                  const float* __restrict__ bb, const float* __restrict__ stats,
                                  int O, int rows, _Float16* __restrict__ a) {
  size_t i = (size_t)blockIdx.x * blockDim.x + threadIdx.x;
  size_t tot = (size_t)rows * O;
  if (i >= tot) return;
  int o = (int)(i % O);
  float cinv = 1.f / (float)rows;
  float mean = stats[o] * cinv;
  float var = stats[O + o] * cinv - mean * mean;
  float t = g[o] * (y[i] - mean) * rsqrtf(var + 1e-5f) + bb[o];
  a[i] = (_Float16)(t >= 0.f ? t : 0.2f * t);
}
__global__ void final_kernel(const float* __restrict__ y, const float* __restrict__ g,
                             const float* __restrict__ bb, const float* __restrict__ stats,
                             float* __restrict__ out) {
  size_t i = (size_t)blockIdx.x * blockDim.x + threadIdx.x;
  size_t tot = (size_t)B_ * 50 * NPTS;
  if (i >= tot) return;
  int n = (int)(i % NPTS);
  int o = (int)((i / NPTS) % 50);
  int b = (int)(i / (50 * NPTS));
  float cinv = 1.f / 32768.f;
  float mean = stats[o] * cinv;
  float var = stats[50 + o] * cinv - mean * mean;
  float t = g[o] * (y[((size_t)b * NPTS + n) * 64 + o] - mean) * rsqrtf(var + 1e-5f) + bb[o];
  out[i] = t >= 0.f ? t : 0.2f * t;
}

// --------------------------- host-side launcher ----------------------------
static inline unsigned cdivu(size_t a, size_t b) { return (unsigned)((a + b - 1) / b); }

extern "C" void kernel_launch(void* const* d_in, const int* in_sizes, int n_in,
                              void* d_out, int out_size, void* d_ws, size_t ws_size,
                              hipStream_t stream) {
  (void)in_sizes; (void)n_in; (void)out_size; (void)ws_size;
  const float* data  = (const float*)d_in[0];
  const float* l_in  = (const float*)d_in[1];
  const float* dw1   = (const float*)d_in[2];
  const float* dg1   = (const float*)d_in[3];
  const float* db1   = (const float*)d_in[4];
  const float* dw2   = (const float*)d_in[5];
  const float* dg2   = (const float*)d_in[6];
  const float* db2   = (const float*)d_in[7];
  const float* dw3   = (const float*)d_in[8];
  const float* dg3   = (const float*)d_in[9];
  const float* db3   = (const float*)d_in[10];
  const float* lin_w = (const float*)d_in[11];
  const float* ln_w  = (const float*)d_in[12];
  const float* ln_b  = (const float*)d_in[13];
  const float* capsW = (const float*)d_in[14];
  const float* cw1   = (const float*)d_in[15];
  const float* cg1   = (const float*)d_in[16];
  const float* cb1   = (const float*)d_in[17];
  const float* cw2   = (const float*)d_in[18];
  const float* cg2   = (const float*)d_in[19];
  const float* cb2   = (const float*)d_in[20];
  const float* cw3   = (const float*)d_in[21];
  const float* cw4   = (const float*)d_in[22];
  const float* cg4   = (const float*)d_in[23];
  const float* cb4   = (const float*)d_in[24];
  const float* cw5   = (const float*)d_in[25];
  const float* cg5   = (const float*)d_in[26];
  const float* cb5   = (const float*)d_in[27];
  float* out = (float*)d_out;

  char* ws = (char*)d_ws;
  size_t off = 0;
  auto alloc = [&](size_t bytes) -> char* {
    char* p = ws + off;
    off += (bytes + 255) & ~(size_t)255;
    return p;
  };

  int*       idx    = (int*)alloc((size_t)B_ * NPTS * KNNK * 4);
  float*     x1     = (float*)alloc((size_t)B_ * 64 * NPTS * 4);
  float*     x2     = (float*)alloc((size_t)B_ * 64 * NPTS * 4);
  float*     x3     = (float*)alloc((size_t)B_ * 128 * NPTS * 4);
  unsigned*  ymax   = (unsigned*)alloc((size_t)B_ * 128 * NPTS * 4);
  unsigned*  ymin   = (unsigned*)alloc((size_t)B_ * 128 * NPTS * 4);
  float*     stats  = (float*)alloc(512 * 4);
  _Float16*  wdw1   = (_Float16*)alloc(64 * 32 * 2);
  _Float16*  wdw2   = (_Float16*)alloc(64 * 128 * 2);
  _Float16*  wdw3   = (_Float16*)alloc(128 * 128 * 2);
  _Float16*  wlin   = (_Float16*)alloc(48 * 256 * 2);    // rows padded 40->48
  _Float16*  wcw2   = (_Float16*)alloc(256 * 1152 * 2);
  _Float16*  wcw3   = (_Float16*)alloc(256 * 256 * 2);
  _Float16*  wcw4   = (_Float16*)alloc(128 * 256 * 2);
  _Float16*  wcw5   = (_Float16*)alloc(64 * 128 * 2);    // rows padded 50->64
  _Float16*  feat16 = (_Float16*)alloc((size_t)B_ * NPTS * 256 * 2);
  float*     h      = (float*)alloc((size_t)B_ * NPTS * 40 * 4);
  float*     lnst   = (float*)alloc(64 * 4);
  float*     enc32  = (float*)alloc((size_t)B_ * NPTS * 40 * 4);
  _Float16*  enc16  = (_Float16*)alloc((size_t)B_ * NPTS * 64 * 2);
  float*     bij    = (float*)alloc((size_t)B_ * 40 * NPTS * 4);
  float*     cc     = (float*)alloc((size_t)B_ * 40 * NPTS * 4);
  float*     vout   = (float*)alloc((size_t)B_ * 40 * 64 * 4);
  _Float16*  latT   = (_Float16*)alloc((size_t)B_ * 64 * 64 * 2);
  float*     embf   = (float*)alloc((size_t)B_ * NPTS * 64 * 4);
  float*     embmx  = (float*)alloc((size_t)B_ * NPTS * 4);
  float*     labact = (float*)alloc((size_t)B_ * 64 * 4);
  float*     uhat   = (float*)alloc((size_t)B_ * 40 * NPTS * 64 * 4);
  char*      capsrg = alloc((size_t)40 * NPTS * 64 * 48 * 2 + 1024);  // +slack for f16 tail overread
  _Float16*  capsw16 = (_Float16*)capsrg;
  // decoder buffers overlay the capsW-f16 region (last used by uhat_kernel)
  _Float16*  embfull = (_Float16*)capsrg;
  float*     ya      = (float*)(capsrg + ((size_t)B_ * NPTS * 1152 * 2));
  float*     yb      = (float*)((char*)ya + (size_t)B_ * NPTS * 256 * 4);
  _Float16*  aa      = (_Float16*)((char*)yb + (size_t)B_ * NPTS * 256 * 4);
  _Float16*  ab      = (_Float16*)((char*)aa + (size_t)B_ * NPTS * 256 * 2);

  const int TPB = 256;
  const float invEC = 1.f / (float)((size_t)B_ * NPTS * KNNK);

  // ---- weight conversions (f32 -> f16, rows/cols padded for WMMA tiles) ----
  cvt_pad_kernel<<<cdivu(64 * 32, TPB), TPB, 0, stream>>>(dw1, wdw1, 64, 64, 6, 32);
  cvt_pad_kernel<<<cdivu(64 * 128, TPB), TPB, 0, stream>>>(dw2, wdw2, 64, 64, 128, 128);
  cvt_pad_kernel<<<cdivu(128 * 128, TPB), TPB, 0, stream>>>(dw3, wdw3, 128, 128, 128, 128);
  cvt_pad_kernel<<<cdivu(48 * 256, TPB), TPB, 0, stream>>>(lin_w, wlin, 40, 48, 256, 256);
  cvt_pad_kernel<<<cdivu(256 * 1152, TPB), TPB, 0, stream>>>(cw2, wcw2, 256, 256, 1128, 1152);
  cvt_pad_kernel<<<cdivu(256 * 256, TPB), TPB, 0, stream>>>(cw3, wcw3, 256, 256, 256, 256);
  cvt_pad_kernel<<<cdivu(128 * 256, TPB), TPB, 0, stream>>>(cw4, wcw4, 128, 128, 256, 256);
  cvt_pad_kernel<<<cdivu(64 * 128, TPB), TPB, 0, stream>>>(cw5, wcw5, 50, 64, 128, 128);
  cvt_pad_kernel<<<cdivu((size_t)2621440 * 48, TPB), TPB, 0, stream>>>(capsW, capsw16, 2621440, 2621440, 40, 48);

  dim3 kgrid(NPTS / 8, B_);
  dim3 egrid(NPTS * KNNK / 64, B_);

  // ---- EdgeConv block 1 (C=3 -> 64) ----
  knn_kernel<3><<<kgrid, 256, 0, stream>>>(data, idx);
  fill_u32_kernel<<<cdivu((size_t)B_ * 64 * NPTS, TPB), TPB, 0, stream>>>(ymax, 0u, (size_t)B_ * 64 * NPTS);
  fill_u32_kernel<<<cdivu((size_t)B_ * 64 * NPTS, TPB), TPB, 0, stream>>>(ymin, 0xFFFFFFFFu, (size_t)B_ * 64 * NPTS);
  fill_f32_kernel<<<1, 256, 0, stream>>>(stats, 0.f, 512);
  edgeconv_kernel<3, 32, 64><<<egrid, 128, 0, stream>>>(data, idx, wdw1, ymax, ymin, stats);
  bn_max_kernel<<<cdivu((size_t)B_ * 64 * NPTS, TPB), TPB, 0, stream>>>(ymax, ymin, stats, dg1, db1, x1, 64, invEC);

  // ---- EdgeConv block 2 (C=64 -> 64) ----
  knn_kernel<64><<<kgrid, 256, 0, stream>>>(x1, idx);
  fill_u32_kernel<<<cdivu((size_t)B_ * 64 * NPTS, TPB), TPB, 0, stream>>>(ymax, 0u, (size_t)B_ * 64 * NPTS);
  fill_u32_kernel<<<cdivu((size_t)B_ * 64 * NPTS, TPB), TPB, 0, stream>>>(ymin, 0xFFFFFFFFu, (size_t)B_ * 64 * NPTS);
  fill_f32_kernel<<<1, 256, 0, stream>>>(stats, 0.f, 512);
  edgeconv_kernel<64, 128, 64><<<egrid, 128, 0, stream>>>(x1, idx, wdw2, ymax, ymin, stats);
  bn_max_kernel<<<cdivu((size_t)B_ * 64 * NPTS, TPB), TPB, 0, stream>>>(ymax, ymin, stats, dg2, db2, x2, 64, invEC);

  // ---- EdgeConv block 3 (C=64 -> 128) ----
  knn_kernel<64><<<kgrid, 256, 0, stream>>>(x2, idx);
  fill_u32_kernel<<<cdivu((size_t)B_ * 128 * NPTS, TPB), TPB, 0, stream>>>(ymax, 0u, (size_t)B_ * 128 * NPTS);
  fill_u32_kernel<<<cdivu((size_t)B_ * 128 * NPTS, TPB), TPB, 0, stream>>>(ymin, 0xFFFFFFFFu, (size_t)B_ * 128 * NPTS);
  fill_f32_kernel<<<1, 256, 0, stream>>>(stats, 0.f, 512);
  edgeconv_kernel<64, 128, 128><<<egrid, 128, 0, stream>>>(x2, idx, wdw3, ymax, ymin, stats);
  bn_max_kernel<<<cdivu((size_t)B_ * 128 * NPTS, TPB), TPB, 0, stream>>>(ymax, ymin, stats, dg3, db3, x3, 128, invEC);

  // ---- encoder: feat -> linear -> LayerNorm -> ReLU ----
  feat_kernel<<<cdivu((size_t)B_ * NPTS * 256, TPB), TPB, 0, stream>>>(x1, x2, x3, feat16);
  gemm_f16_kernel<<<dim3(4, 3, B_), 128, 0, stream>>>(wlin, 0, feat16, (size_t)NPTS * 256,
                                                      h, (size_t)NPTS * 40, 40, 256, NPTS, 40);
  ln_stats_kernel<<<B_, 256, 0, stream>>>(h, lnst);
  enc_kernel<<<cdivu((size_t)B_ * NPTS * 64, TPB), TPB, 0, stream>>>(h, ln_w, ln_b, lnst, enc32, enc16);

  // ---- capsule projection + dynamic routing (3 iters) ----
  uhat_kernel<<<dim3(NPTS, 40), 128, 0, stream>>>(enc16, capsw16, uhat);
  fill_f32_kernel<<<cdivu((size_t)B_ * 40 * NPTS, TPB), TPB, 0, stream>>>(bij, 0.f, (size_t)B_ * 40 * NPTS);
  for (int it = 0; it < 3; ++it) {
    softmax_l_kernel<<<cdivu((size_t)B_ * NPTS, TPB), TPB, 0, stream>>>(bij, cc);
    sv_squash_kernel<<<dim3(40, B_), 256, 0, stream>>>(cc, uhat, vout);
    if (it < 2)
      bij_update_kernel<<<cdivu((size_t)B_ * 40 * NPTS, TPB), TPB, 0, stream>>>(vout, uhat, bij);
  }

  // ---- emb = enc x latent, max over v; lab branch; assemble decoder input ----
  latT_kernel<<<cdivu((size_t)B_ * 64 * 64, TPB), TPB, 0, stream>>>(vout, latT);
  gemm_f16_kernel<<<dim3(4, 4, B_), 128, 0, stream>>>(latT, (size_t)64 * 64, enc16, (size_t)NPTS * 64,
                                                      embf, (size_t)NPTS * 64, 64, 64, NPTS, 64);
  embmax_kernel<<<cdivu((size_t)B_ * NPTS, TPB), TPB, 0, stream>>>(embf, embmx);
  lab_kernel<<<1, 64, 0, stream>>>(l_in, cw1, cg1, cb1, labact);
  embfull_kernel<<<cdivu((size_t)B_ * NPTS * 1152, TPB), TPB, 0, stream>>>(embmx, labact, enc32, embfull);

  // ---- decoder: 1128->256->256->128->50 (GEMM + BN + LeakyReLU) ----
  gemm_f16_kernel<<<dim3(4, 16, B_), 128, 0, stream>>>(wcw2, 0, embfull, (size_t)NPTS * 1152,
                                                       ya, (size_t)NPTS * 256, 256, 1152, NPTS, 256);
  chan_stats_kernel<<<256, 256, 0, stream>>>(ya, 256, 256, B_ * NPTS, stats);
  bn_act_f16_kernel<<<cdivu((size_t)B_ * NPTS * 256, TPB), TPB, 0, stream>>>(ya, cg2, cb2, stats, 256, B_ * NPTS, aa);

  gemm_f16_kernel<<<dim3(4, 16, B_), 128, 0, stream>>>(wcw3, 0, aa, (size_t)NPTS * 256,
                                                       yb, (size_t)NPTS * 256, 256, 256, NPTS, 256);
  chan_stats_kernel<<<256, 256, 0, stream>>>(yb, 256, 256, B_ * NPTS, stats);
  bn_act_f16_kernel<<<cdivu((size_t)B_ * NPTS * 256, TPB), TPB, 0, stream>>>(yb, cg2, cb2, stats, 256, B_ * NPTS, ab);

  gemm_f16_kernel<<<dim3(4, 8, B_), 128, 0, stream>>>(wcw4, 0, ab, (size_t)NPTS * 256,
                                                      ya, (size_t)NPTS * 128, 128, 256, NPTS, 128);
  chan_stats_kernel<<<128, 256, 0, stream>>>(ya, 128, 128, B_ * NPTS, stats);
  bn_act_f16_kernel<<<cdivu((size_t)B_ * NPTS * 128, TPB), TPB, 0, stream>>>(ya, cg4, cb4, stats, 128, B_ * NPTS, aa);

  gemm_f16_kernel<<<dim3(4, 4, B_), 128, 0, stream>>>(wcw5, 0, aa, (size_t)NPTS * 128,
                                                      yb, (size_t)NPTS * 64, 50, 128, NPTS, 64);
  chan_stats_kernel<<<50, 256, 0, stream>>>(yb, 64, 50, B_ * NPTS, stats);
  final_kernel<<<cdivu((size_t)B_ * 50 * NPTS, TPB), TPB, 0, stream>>>(yb, cg5, cb5, stats, out);
}
